// NeuralSheet_57878979281561
// MI455X (gfx1250) — compile-verified, matrix operand order checked by reading, not run on hardware
//
#include <hip/hip_runtime.h>
#include <hip/hip_bf16.h>
#include <math.h>

// ---------------------------------------------------------------------------
// Problem constants
// ---------------------------------------------------------------------------
#define S 64
#define N 4096            // S*S
#define RF 15
#define INPUT_SIZE 96
#define ITERS 30
#define PARTS 8           // K-partitions per matvec (parallelism)
#define KB_TOTAL 128      // 4096 / 32 k-chunks
#define KB_PER_PART (KB_TOTAL / PARTS)   // 16

typedef __attribute__((ext_vector_type(16))) __bf16 v16bf;
typedef __attribute__((ext_vector_type(8)))  __bf16 v8bf;
typedef __attribute__((ext_vector_type(8)))  float  v8f;

// Workspace layout (bytes), all 256-aligned
#define WP_OFF   ((size_t)0)                       // packed W, bf16: N*N*2 = 33,554,432
#define AFF_OFF  ((size_t)33554432)                // aff  f32: 16384
#define RC_OFF   (AFF_OFF + 16384)                 // rc   f32: 16384
#define R4B_OFF  (RC_OFF + 16384)                  // r4   bf16: 8192
#define YP_OFF   (R4B_OFF + 8192)                  // ypart f32: PARTS*N*4 = 131072

// ---------------------------------------------------------------------------
// 1) Afferent drive: gather 15x15 patch per neuron, dot with its weights.
// ---------------------------------------------------------------------------
__global__ void aff_kernel(const float* __restrict__ inp,
                           const int*   __restrict__ rf,
                           const float* __restrict__ aw,
                           float*       __restrict__ aff) {
    int n = blockIdx.x * blockDim.x + threadIdx.x;
    if (n >= N) return;
    int y0 = rf[2 * n + 0];
    int x0 = rf[2 * n + 1];
    const float* w = aw + (size_t)n * (RF * RF);
    float s = 0.f;
    for (int dy = 0; dy < RF; ++dy) {
        const float* row = inp + (y0 + dy) * INPUT_SIZE + x0;
        const float* wr  = w + dy * RF;
        #pragma unroll
        for (int dx = 0; dx < RF; ++dx) s = fmaf(row[dx], wr[dx], s);
    }
    aff[n] = s;
}

// ---------------------------------------------------------------------------
// 2) Build W = lwe - mri (row-normalized), stored bf16 in WMMA A-fragment
//    layout so the matvec can load fragments with contiguous b128 loads.
//    16-bit A 16x32 layout: lane = m + 16*((k&15)>=8), slot = (k&7) + 8*(k>=16)
//    One 256-thread workgroup per matrix row i.
// ---------------------------------------------------------------------------
__global__ void pack_kernel(const float* __restrict__ lwe,
                            const float* __restrict__ l4c,
                            const float* __restrict__ masks,
                            const float* __restrict__ eye,
                            __bf16*      __restrict__ Wp) {
    const int i = blockIdx.x;      // row 0..4095
    const int t = threadIdx.x;     // 0..255
    const size_t roff = (size_t)i * N;

    __shared__ float red[256];
    float s = 0.f;
    for (int j = t; j < N; j += 256) {
        float m = l4c[roff + j] * (1.f - masks[roff + j]) * eye[roff + j];
        s += m;
    }
    red[t] = s;
    __syncthreads();
    for (int w = 128; w > 0; w >>= 1) {
        if (t < w) red[t] += red[t + w];
        __syncthreads();
    }
    const float inv = 1.f / (red[0] + 1e-11f);

    const int rb  = i >> 4;        // 16-row block
    const int m16 = i & 15;        // M within block
    for (int jp = t; jp < N / 2; jp += 256) {
        int j0 = jp * 2;           // even j; (j0, j0+1) land in adjacent slots
        float m0 = l4c[roff + j0]     * (1.f - masks[roff + j0])     * eye[roff + j0];
        float m1 = l4c[roff + j0 + 1] * (1.f - masks[roff + j0 + 1]) * eye[roff + j0 + 1];
        float w0 = lwe[roff + j0]     - m0 * inv;
        float w1 = lwe[roff + j0 + 1] - m1 * inv;
        int kb   = j0 >> 5;
        int k    = j0 & 31;        // even
        int lane = m16 + (((k & 15) >= 8) ? 16 : 0);
        int slot = (k & 7) + ((k >= 16) ? 8 : 0);
        size_t idx = ((((size_t)rb * KB_TOTAL + kb) * 32 + lane) * 16 + slot);
        union { __bf16 h[2]; unsigned int u; } p;
        p.h[0] = (__bf16)w0;
        p.h[1] = (__bf16)w1;
        *reinterpret_cast<unsigned int*>(Wp + idx) = p.u;   // 4B aligned (slot even)
    }
}

// ---------------------------------------------------------------------------
// 3) State init: r4 = 0 (bf16), rc = 0
// ---------------------------------------------------------------------------
__global__ void init_kernel(__bf16* __restrict__ r4b, float* __restrict__ rc) {
    int idx = blockIdx.x * blockDim.x + threadIdx.x;
    if (idx < N) {
        r4b[idx] = (__bf16)0.f;
        rc[idx]  = 0.f;
    }
}

// ---------------------------------------------------------------------------
// 4) Matvec y_part = W_part @ r4 via v_wmma_f32_16x16x32_bf16.
//    2048 waves: wave -> (rb = wave&255 : 16 output rows, part = wave>>8).
//    B replicates the x-chunk across all 16 columns (layout: lanes 0-15 hold
//    K 0-7 / 16-23; lanes 16-31 hold K 8-15 / 24-31), so every D column holds
//    the same partial dot; lanes N==0 write the 16 results.
// ---------------------------------------------------------------------------
__global__ void __launch_bounds__(256)
matvec_kernel(const __bf16* __restrict__ Wp,
              const __bf16* __restrict__ xb,
              float*        __restrict__ ypart) {
    const int wave = (blockIdx.x * blockDim.x + threadIdx.x) >> 5;
    const int lane = threadIdx.x & 31;
    const int rb   = wave & 255;
    const int part = wave >> 8;          // 0..PARTS-1
    const int half = lane >> 4;          // 0 | 1

    v8f acc = {};
    const __bf16* A  = Wp + ((size_t)rb * KB_TOTAL + part * KB_PER_PART) * 512 + lane * 16;
    const __bf16* x0 = xb + part * KB_PER_PART * 32;

    #pragma unroll 4
    for (int kb = 0; kb < KB_PER_PART; ++kb) {
        v16bf a = *(const v16bf*)(A + kb * 512);            // 32B contiguous per lane
        const __bf16* xc = x0 + kb * 32 + half * 8;
        v8bf blo = *(const v8bf*)(xc);                      // K base+half*8 .. +7
        v8bf bhi = *(const v8bf*)(xc + 16);                 // K base+16+half*8 .. +7
        v16bf b;
        #pragma unroll
        for (int e = 0; e < 8; ++e) { b[e] = blo[e]; b[e + 8] = bhi[e]; }
        acc = __builtin_amdgcn_wmma_f32_16x16x32_bf16(
                  /*neg_a=*/false, a, /*neg_b=*/false, b,
                  /*c_mod=*/(short)0, acc, /*reuse_a=*/false, /*reuse_b=*/false);
    }

    if ((lane & 15) == 0) {              // N == 0 column: lane 0 -> M 0..7, lane 16 -> M 8..15
        float* yp = ypart + part * N + rb * 16 + (half << 3);
        #pragma unroll
        for (int v = 0; v < 8; ++v) yp[v] = acc[v];
    }
}

// ---------------------------------------------------------------------------
// 5) Nonlinearity + state update. Writes r4 as bf16 (next matvec input) and
//    fp32 into d_out (last iteration leaves the final answer in place).
// ---------------------------------------------------------------------------
__global__ void update_kernel(const float* __restrict__ ypart,
                              const float* __restrict__ aff,
                              const float* __restrict__ th,
                              float*       __restrict__ rc,
                              __bf16*      __restrict__ r4b,
                              float*       __restrict__ out) {
    int n = blockIdx.x * blockDim.x + threadIdx.x;
    if (n >= N) return;
    float y = 0.f;
    #pragma unroll
    for (int p = 0; p < PARTS; ++p) y += ypart[p * N + n];
    float rco = rc[n];
    const float b = 0.5f / (0.5f + 1e-11f);
    float l4aff = aff[n] * 0.9f + rco * b * 0.1f;
    float v = l4aff + y - th[n];
    float r = tanhf(fmaxf(v, 0.f));
    out[n] = r;
    r4b[n] = (__bf16)r;
    rc[n]  = tanhf(rco);
}

// ---------------------------------------------------------------------------
// Launch
// ---------------------------------------------------------------------------
extern "C" void kernel_launch(void* const* d_in, const int* in_sizes, int n_in,
                              void* d_out, int out_size, void* d_ws, size_t ws_size,
                              hipStream_t stream) {
    const float* input = (const float*)d_in[0];   // [1,1,96,96]
    const int*   rf    = (const int*)  d_in[1];   // [N,2]
    const float* aw    = (const float*)d_in[2];   // [N,1,15,15]
    const float* lwe   = (const float*)d_in[3];   // [N,1,64,64]
    const float* l4c   = (const float*)d_in[4];   // [N,1,64,64]
    const float* masks = (const float*)d_in[5];   // [N,1,64,64]
    const float* eye   = (const float*)d_in[6];   // [N,1,64,64]
    const float* th    = (const float*)d_in[7];   // [1,1,64,64]
    float* out = (float*)d_out;                   // [1,1,64,64]

    char* ws = (char*)d_ws;
    __bf16* Wp    = (__bf16*)(ws + WP_OFF);
    float*  aff   = (float*) (ws + AFF_OFF);
    float*  rc    = (float*) (ws + RC_OFF);
    __bf16* r4b   = (__bf16*)(ws + R4B_OFF);
    float*  ypart = (float*) (ws + YP_OFF);

    // Prep (recomputed every call: deterministic, no cached state)
    aff_kernel <<<N / 256, 256, 0, stream>>>(input, rf, aw, aff);
    pack_kernel<<<N,       256, 0, stream>>>(lwe, l4c, masks, eye, Wp);
    init_kernel<<<N / 256, 256, 0, stream>>>(r4b, rc);

    // 30 fixed-point iterations: bf16-WMMA matvec (L2-resident W) + pointwise
    for (int it = 0; it < ITERS; ++it) {
        matvec_kernel<<<(256 * PARTS * 32) / 256, 256, 0, stream>>>(Wp, r4b, ypart);
        update_kernel<<<N / 256, 256, 0, stream>>>(ypart, aff, th, rc, r4b, out);
    }
}
